// CATS_Attention_70300024701825
// MI455X (gfx1250) — compile-verified, matrix operand order checked by reading, not run on hardware
//
#include <hip/hip_runtime.h>
#include <hip/hip_bf16.h>

#define EMB   768
#define NN    128
#define MM    1024
#define XROWS (3*EMB + 2)   // 2306

typedef __attribute__((ext_vector_type(16))) _Float16 v16h;
typedef __attribute__((ext_vector_type(8)))  _Float16 v8h;
typedef __attribute__((ext_vector_type(2)))  __fp16   v2fp16;  // matches cvt_pkrtz return
typedef __attribute__((ext_vector_type(8)))  float    v8f;

union V16 { v16h v; v8h h[2]; };
union H2  { v2fp16 h; unsigned u; };

// Fast tanh from native v_exp_f32 + v_rcp_f32.
__device__ __forceinline__ float fast_tanh(float x) {
    float e = __expf(2.0f * x);
    float r = __builtin_amdgcn_rcpf(e + 1.0f);
    return (e - 1.0f) * r;
}

// ---------------------------------------------------------------------------
// Kernel 0: convert reused weights fp32 -> fp16 (tiny; weights stay in L2)
// ---------------------------------------------------------------------------
__global__ void cvt_kernel(const float* __restrict__ Wa,
                           const float* __restrict__ W1,
                           const float* __restrict__ W2,
                           _Float16* __restrict__ WaH,
                           _Float16* __restrict__ W1H,
                           _Float16* __restrict__ W2H) {
    int idx = blockIdx.x * blockDim.x + threadIdx.x;
    int stride = gridDim.x * blockDim.x;
    for (int i = idx; i < 128 * 2 * EMB; i += stride) WaH[i] = (_Float16)Wa[i];
    for (int i = idx; i < EMB * EMB;     i += stride) W1H[i] = (_Float16)W1[i];
    for (int i = idx; i < EMB * EMB;     i += stride) W2H[i] = (_Float16)W2[i];
}

// ---------------------------------------------------------------------------
// Kernel 1: qv[m,e] = X[m, e, 0] -> f16 rows 2048..3071 of the MLP input
// ---------------------------------------------------------------------------
__global__ void qv_kernel(const float* __restrict__ X, _Float16* __restrict__ xH) {
    const int m = blockIdx.x;
    const float* Xm = X + (size_t)m * XROWS * NN;
    _Float16* out = xH + (size_t)(2 * MM + m) * EMB;
    for (int e = threadIdx.x; e < EMB; e += blockDim.x)
        out[e] = (_Float16)Xm[(size_t)e * NN];
}

// ---------------------------------------------------------------------------
// One 128x768x128 matmul phase: stage f32 rows -> f16 LDS (transposed [n][k]),
// then 8 waves x 8 tiles of v_wmma_f32_16x16x32_f16 accumulate into acc.
//   srcRows : 768 consecutive f32 rows (stride NN) of X
//   waBase  : WaH + arow*(2*EMB) + aoff  (per-thread A-fragment row base)
// ---------------------------------------------------------------------------
__device__ __forceinline__ void mm_phase(const float* __restrict__ srcRows,
                                         const _Float16* __restrict__ waBase,
                                         v8f acc[8], _Float16* Bs,
                                         int sn, int kh, int lane, int kbase) {
    for (int e0 = 0; e0 < EMB; e0 += 32) {
        // Stage 32 rows x 128 cols to LDS f16, transposed to [n][k]:
        // 16 coalesced dword loads, 8 packed cvt, 2 x ds_store_b128 / thread.
        {
            const float* src = srcRows + (size_t)(e0 + kh) * NN + sn;
            unsigned pk[8];
            for (int p = 0; p < 8; ++p) {
                float x0 = src[(size_t)(2 * p) * NN];
                float x1 = src[(size_t)(2 * p + 1) * NN];
                H2 h2; h2.h = __builtin_amdgcn_cvt_pkrtz(x0, x1);
                pk[p] = h2.u;
            }
            uint4* dst = (uint4*)&Bs[sn * 32 + kh];
            dst[0] = make_uint4(pk[0], pk[1], pk[2], pk[3]);
            dst[1] = make_uint4(pk[4], pk[5], pk[6], pk[7]);
        }
        __syncthreads();

        V16 a;
        const _Float16* ap = waBase + e0;
        a.h[0] = *(const v8h*)(ap + kbase);
        a.h[1] = *(const v8h*)(ap + 16 + kbase);

        for (int tile = 0; tile < 8; ++tile) {
            int bn = tile * 16 + (lane & 15);
            const _Float16* bp = &Bs[bn * 32];
            V16 b;
            b.h[0] = *(const v8h*)(bp + kbase);
            b.h[1] = *(const v8h*)(bp + 16 + kbase);
            acc[tile] = __builtin_amdgcn_wmma_f32_16x16x32_f16(
                false, a.v, false, b.v, (short)0, acc[tile], false, false);
        }
        __syncthreads();
    }
}

// ---------------------------------------------------------------------------
// Kernel 2: both pools for one m per block.
//   Phase A: accC  = Wa[:, :768] @ Xq        (shared by both pools)
//   Phase B: accP  = accC + Wa[:, 768:] @ Xp1 -> score1/softmax1/pool1
//   Phase C: accC += Wa[:, 768:] @ Xp2        -> score2/softmax2/pool2
// Saves 25% of attention WMMA work and reads Xq from HBM once, not twice.
// ---------------------------------------------------------------------------
__global__ __launch_bounds__(256) void attn_kernel(const float* __restrict__ X,
                                                   const float* __restrict__ va,
                                                   const _Float16* __restrict__ WaH,
                                                   _Float16* __restrict__ xH) {
    const int m    = blockIdx.x;
    const int t    = threadIdx.x;
    const int wave = t >> 5;
    const int lane = t & 31;

    __shared__ _Float16 Bs[NN * 32];
    __shared__ float scoreS[NN];
    __shared__ float betaS[NN];
    __shared__ float vaS[NN];

    const float* Xm = X + (size_t)m * XROWS * NN;

    if (t < NN) { vaS[t] = va[t]; scoreS[t] = 0.0f; }
    __syncthreads();

    const int arow  = wave * 16 + (lane & 15);
    const int kbase = (lane >> 4) * 8;
    const int sn = t & 127;
    const int kh = (t >> 7) << 4;          // 0 or 16
    const int mbase = wave * 16 + (lane >> 4) * 8;

    const _Float16* waQ = WaH + (size_t)arow * (2 * EMB);        // K 0..767
    const _Float16* waP = waQ + EMB;                             // K 768..1535

    v8f accC[8] = {};
    v8f accP[8];

    // Phase A: common Xq product
    mm_phase(Xm, waQ, accC, Bs, sn, kh, lane, kbase);

    for (int pool = 0; pool < 2; ++pool) {
        const int prow0 = (pool == 0) ? EMB       : (2 * EMB + 1);
        const int vrow  = (pool == 0) ? (2 * EMB) : (3 * EMB + 1);
        v8f* acc;
        if (pool == 0) {
            for (int i = 0; i < 8; ++i) accP[i] = accC[i];
            acc = accP;                                    // keep accC for pool 1
        } else {
            acc = accC;                                    // consume accC in place
        }

        // Phase B/C: pool-specific Xp product
        mm_phase(Xm + (size_t)prow0 * NN, waP, acc, Bs, sn, kh, lane, kbase);

        // score[n] += sum_k va[k] * tanh(h[k,n]) over this wave's 16 rows
        for (int tile = 0; tile < 8; ++tile) {
            float part = 0.0f;
            for (int r = 0; r < 8; ++r)
                part += vaS[mbase + r] * fast_tanh(acc[tile][r]);
            part += __shfl_xor(part, 16, 32);
            if (lane < 16) atomicAdd(&scoreS[tile * 16 + lane], part);
        }
        __syncthreads();

        // masked softmax over n (wave 0)
        if (wave == 0) {
            float s[4];
            for (int j = 0; j < 4; ++j) {
                int n = lane + 32 * j;
                s[j] = Xm[(size_t)vrow * NN + n] * scoreS[n];
            }
            float mx = fmaxf(fmaxf(s[0], s[1]), fmaxf(s[2], s[3]));
            for (int o = 16; o > 0; o >>= 1) mx = fmaxf(mx, __shfl_xor(mx, o, 32));
            float sum = 0.0f;
            for (int j = 0; j < 4; ++j) { s[j] = __expf(s[j] - mx); sum += s[j]; }
            for (int o = 16; o > 0; o >>= 1) sum += __shfl_xor(sum, o, 32);
            float inv = __builtin_amdgcn_rcpf(sum);
            for (int j = 0; j < 4; ++j) betaS[lane + 32 * j] = s[j] * inv;
        }
        __syncthreads();

        // p[e] = sum_n beta[n] * Xp[e,n]  (Xp rows are L2-hot from the matmul)
        _Float16* out = xH + ((size_t)pool * MM + m) * EMB;
        const float b0 = betaS[lane * 4 + 0], b1 = betaS[lane * 4 + 1];
        const float b2 = betaS[lane * 4 + 2], b3 = betaS[lane * 4 + 3];
        for (int e = wave; e < EMB; e += 8) {
            const float4 xv = *(const float4*)(Xm + (size_t)(prow0 + e) * NN + lane * 4);
            float d = xv.x * b0 + xv.y * b1 + xv.z * b2 + xv.w * b3;
            for (int o = 16; o > 0; o >>= 1) d += __shfl_xor(d, o, 32);
            if (lane == 0) out[e] = (_Float16)d;
        }
        __syncthreads();

        // reset score accumulator for the second pool
        if (pool == 0) {
            if (t < NN) scoreS[t] = 0.0f;
            __syncthreads();
        }
    }
}

// ---------------------------------------------------------------------------
// Kernel 3/4: C[i,j] = act(sum_e A[i,e]*Bw[j,e] + bias[j]), A:(3072,768) f16,
// Bw:(768,768) f16 row-major. Block = 128x128 tile, 8 waves x 8 n-tiles.
// ---------------------------------------------------------------------------
__global__ __launch_bounds__(256) void mlp_gemm(const _Float16* __restrict__ A,
                                                const _Float16* __restrict__ Bw,
                                                const float* __restrict__ bias,
                                                _Float16* __restrict__ out,
                                                int relu) {
    const int i0 = blockIdx.x * 128;
    const int j0 = blockIdx.y * 128;
    const int wave = threadIdx.x >> 5;
    const int lane = threadIdx.x & 31;
    const int kbase = (lane >> 4) * 8;

    v8f acc[8] = {};
    const int arow = i0 + wave * 16 + (lane & 15);

    for (int e0 = 0; e0 < EMB; e0 += 32) {
        V16 a;
        const _Float16* ap = A + (size_t)arow * EMB + e0;
        a.h[0] = *(const v8h*)(ap + kbase);
        a.h[1] = *(const v8h*)(ap + 16 + kbase);
        for (int tile = 0; tile < 8; ++tile) {
            int j = j0 + tile * 16 + (lane & 15);
            const _Float16* bp = Bw + (size_t)j * EMB + e0;
            V16 b;
            b.h[0] = *(const v8h*)(bp + kbase);
            b.h[1] = *(const v8h*)(bp + 16 + kbase);
            acc[tile] = __builtin_amdgcn_wmma_f32_16x16x32_f16(
                false, a.v, false, b.v, (short)0, acc[tile], false, false);
        }
    }

    const int ibase = i0 + wave * 16 + (lane >> 4) * 8;
    for (int tile = 0; tile < 8; ++tile) {
        int j = j0 + tile * 16 + (lane & 15);
        float bz = bias[j];
        for (int r = 0; r < 8; ++r) {
            float c = acc[tile][r] + bz;
            if (relu) c = fmaxf(c, 0.0f);
            out[(size_t)(ibase + r) * EMB + j] = (_Float16)c;
        }
    }
}

// ---------------------------------------------------------------------------
// Kernel 5: o[m] = relu(b3 + W3 . [z1, z2, |z1-z2|, |z1-zq|, |z2-zq|])
// ---------------------------------------------------------------------------
__global__ __launch_bounds__(256) void final_kernel(const _Float16* __restrict__ zH,
                                                    const float* __restrict__ W3,
                                                    const float* __restrict__ b3,
                                                    float* __restrict__ out) {
    const int m = blockIdx.x;
    __shared__ float red[8];
    const _Float16* z1 = zH + (size_t)m * EMB;
    const _Float16* z2 = zH + (size_t)(MM + m) * EMB;
    const _Float16* zq = zH + (size_t)(2 * MM + m) * EMB;
    float acc = 0.0f;
    for (int e = threadIdx.x; e < EMB; e += 256) {
        float a = (float)z1[e], b = (float)z2[e], c = (float)zq[e];
        acc += W3[e] * a + W3[EMB + e] * b + W3[2 * EMB + e] * fabsf(a - b)
             + W3[3 * EMB + e] * fabsf(a - c) + W3[4 * EMB + e] * fabsf(b - c);
    }
    for (int o = 16; o > 0; o >>= 1) acc += __shfl_xor(acc, o, 32);
    const int wave = threadIdx.x >> 5, lane = threadIdx.x & 31;
    if (lane == 0) red[wave] = acc;
    __syncthreads();
    if (threadIdx.x == 0) {
        float s = 0.0f;
        for (int w = 0; w < 8; ++w) s += red[w];
        out[m] = fmaxf(s + b3[0], 0.0f);
    }
}

// ---------------------------------------------------------------------------
extern "C" void kernel_launch(void* const* d_in, const int* in_sizes, int n_in,
                              void* d_out, int out_size, void* d_ws, size_t ws_size,
                              hipStream_t stream) {
    (void)in_sizes; (void)n_in; (void)out_size; (void)ws_size;
    const float* X  = (const float*)d_in[0];
    const float* Wa = (const float*)d_in[1];
    const float* va = (const float*)d_in[2];
    const float* W1 = (const float*)d_in[3];
    const float* b1 = (const float*)d_in[4];
    const float* W2 = (const float*)d_in[5];
    const float* b2 = (const float*)d_in[6];
    const float* W3 = (const float*)d_in[7];
    const float* b3 = (const float*)d_in[8];

    char* ws = (char*)d_ws;
    _Float16* WaH = (_Float16*)(ws);                       //  0.39 MB
    _Float16* W1H = (_Float16*)(ws + 393216);              //  1.18 MB
    _Float16* W2H = (_Float16*)(ws + 1572864);             //  1.18 MB
    _Float16* xH  = (_Float16*)(ws + 2752512);             //  4.72 MB  (3072 x 768 f16)
    _Float16* h1H = (_Float16*)(ws + 7471104);             //  4.72 MB
    _Float16* zH  = (_Float16*)(ws + 12189696);            //  4.72 MB
    float* outp = (float*)d_out;

    cvt_kernel<<<512, 256, 0, stream>>>(Wa, W1, W2, WaH, W1H, W2H);
    qv_kernel<<<MM, 256, 0, stream>>>(X, xH);
    attn_kernel<<<MM, 256, 0, stream>>>(X, va, WaH, xH);
    mlp_gemm<<<dim3(3072 / 128, EMB / 128), 256, 0, stream>>>(xH, W1H, b1, h1H, 0);
    mlp_gemm<<<dim3(3072 / 128, EMB / 128), 256, 0, stream>>>(h1H, W2H, b2, zH, 1);
    final_kernel<<<MM, 256, 0, stream>>>(zH, W3, b3, outp);
}
